// QuantumEmbedding_8555574854205
// MI455X (gfx1250) — matrix operand order repaired and checked by reading, not compiled, not run on hardware
//
#include <hip/hip_runtime.h>
#include <hip/hip_bf16.h>

typedef __attribute__((ext_vector_type(16))) __bf16 v16bf;
typedef __attribute__((ext_vector_type(8)))  __bf16 v8bf;
typedef __attribute__((ext_vector_type(4)))  __bf16 v4bf;
typedef __attribute__((ext_vector_type(8)))  float  v8f;
typedef __attribute__((ext_vector_type(4)))  float  v4f;

#define M_DIM 4096
#define N_DIM 4096
#define K_DIM 1024
#define TILE_M 128
#define TILE_N 128
#define TILE_K 32
#define LDSTR  40   // 32 bf16 + 8 bf16 pad -> 80-byte row stride (conflict spread)

// ---------------------------------------------------------------------------
// Kernel 1: phase_mean[j] = mean(quantum_phases[j, 0:12])
// ---------------------------------------------------------------------------
__global__ void qe_phase_mean(const float* __restrict__ phases,
                              float* __restrict__ pm, int n) {
    int j = blockIdx.x * blockDim.x + threadIdx.x;
    if (j < n) {
        float s = 0.f;
#pragma unroll
        for (int q = 0; q < 12; ++q) s += phases[j * 12 + q];
        pm[j] = s * (1.0f / 12.0f);
    }
}

// ---------------------------------------------------------------------------
// Kernel 2: out = cos(x * pm) @ W   via v_wmma_f32_16x16x32_bf16
// Block: 256 threads (8 waves), 128x128 output tile.
// Wave grid 4(M) x 2(N): each wave -> 32x64 region = 2x4 WMMA accumulators.
// ---------------------------------------------------------------------------
__global__ __launch_bounds__(256) void qe_wmma_gemm(
    const float* __restrict__ x,   // [4096, 1024]
    const float* __restrict__ w,   // [1024, 4096]
    const float* __restrict__ pm,  // [1024]
    float* __restrict__ out)       // [4096, 4096]
{
    __shared__ __bf16 As[TILE_M * LDSTR];  // As[m][k], k in [0,32)
    __shared__ __bf16 Bs[TILE_N * LDSTR];  // Bs[n][k]  (W transposed)

    const int tid  = threadIdx.x;
    const int lane = tid & 31;
    const int wave = tid >> 5;
    const int half = (lane >> 4) & 1;  // lane half (K-split select)
    const int l16  = lane & 15;

    const int block_m = blockIdx.y * TILE_M;
    const int block_n = blockIdx.x * TILE_N;

    const int wm = (wave & 3) * 32;   // wave M offset in tile
    const int wn = (wave >> 2) * 64;  // wave N offset in tile

    v8f acc[2][4];
#pragma unroll
    for (int i = 0; i < 2; ++i)
#pragma unroll
        for (int j = 0; j < 4; ++j) acc[i][j] = (v8f)0.0f;

    // --- staging index map ---
    // A: thread -> (row a_m0 + 32r, float4 group a_c of K)   : coalesced x reads
    const int a_c  = tid & 7;   // K float4 group: 4*a_c .. 4*a_c+3
    const int a_m0 = tid >> 3;  // 0..31
    // B: thread -> (column b_n of W, 4 consecutive K)        : coalesced w reads
    const int b_n  = tid & 127;        // 0..127
    const int b_k0 = (tid >> 7) * 4;   // 0 or 4; + 8r

    for (int k0 = 0; k0 < K_DIM; k0 += TILE_K) {
        __syncthreads();

        // ---- stage A: cos(x * pm) -> bf16 into As[m][k] ----
        {
            const v4f pv = *(const v4f*)(pm + k0 + 4 * a_c);
#pragma unroll
            for (int r = 0; r < 4; ++r) {
                const int m = a_m0 + 32 * r;
                const v4f xv = *(const v4f*)(x + (size_t)(block_m + m) * K_DIM + k0 + 4 * a_c);
                v4bf cv;
#pragma unroll
                for (int q = 0; q < 4; ++q) cv[q] = (__bf16)__cosf(xv[q] * pv[q]);
                *(v4bf*)&As[m * LDSTR + 4 * a_c] = cv;  // 8B store, aligned
            }
        }

        // ---- stage B: W[k][n] -> bf16 into Bs[n][k] (transpose in LDS) ----
#pragma unroll
        for (int r = 0; r < 4; ++r) {
            const int kk = b_k0 + 8 * r;  // 0..31
            v4bf bv;
#pragma unroll
            for (int q = 0; q < 4; ++q)
                bv[q] = (__bf16)w[(size_t)(k0 + kk + q) * N_DIM + block_n + b_n];
            *(v4bf*)&Bs[b_n * LDSTR + kk] = bv;  // 8B store, aligned
        }

        // prefetch next K-tile into cache while we compute (global_prefetch_b8)
        if (k0 + TILE_K < K_DIM) {
            __builtin_prefetch(x + (size_t)(block_m + a_m0) * K_DIM + k0 + TILE_K + 4 * a_c, 0, 0);
            __builtin_prefetch(w + (size_t)(k0 + TILE_K + b_k0) * N_DIM + block_n + b_n, 0, 0);
        }

        __syncthreads();

        // ---- load fragments per ISA 16-bit layouts ----
        // A 16x32: lane l<16 -> M=l, K = {0..7, 16..23}; upper half: K += 8
        v16bf af[2];
#pragma unroll
        for (int i = 0; i < 2; ++i) {
            const __bf16* p = &As[(wm + i * 16 + l16) * LDSTR + 8 * half];
            v8bf lo = *(const v8bf*)p;         // K = 8h .. 8h+7
            v8bf hi = *(const v8bf*)(p + 16);  // K = 8h+16 .. 8h+23
            af[i] = __builtin_shufflevector(lo, hi, 0, 1, 2, 3, 4, 5, 6, 7,
                                                    8, 9, 10, 11, 12, 13, 14, 15);
        }
        // B 32x16: lane -> N = lane&15, K = {8h..8h+7, 8h+16..8h+23} (matches A)
        v16bf bfr[4];
#pragma unroll
        for (int j = 0; j < 4; ++j) {
            const __bf16* p = &Bs[(wn + j * 16 + l16) * LDSTR + 8 * half];
            v8bf lo = *(const v8bf*)p;
            v8bf hi = *(const v8bf*)(p + 16);
            bfr[j] = __builtin_shufflevector(lo, hi, 0, 1, 2, 3, 4, 5, 6, 7,
                                                     8, 9, 10, 11, 12, 13, 14, 15);
        }

        // ---- 8 WMMAs: D = A x B + C ----
#pragma unroll
        for (int i = 0; i < 2; ++i)
#pragma unroll
            for (int j = 0; j < 4; ++j)
                acc[i][j] = __builtin_amdgcn_wmma_f32_16x16x32_bf16(
                    /*neg_a=*/false, af[i], /*neg_b=*/false, bfr[j],
                    /*c_mod=*/(short)0, acc[i][j],
                    /*reuse_a=*/false, /*reuse_b=*/false);
    }

    // ---- store: C/D f32 16x16 layout: lane half selects M+8, VGPR r = M row ----
    float* outp = out + (size_t)(block_m + wm) * N_DIM + block_n + wn;
#pragma unroll
    for (int i = 0; i < 2; ++i)
#pragma unroll
        for (int j = 0; j < 4; ++j)
#pragma unroll
            for (int r = 0; r < 8; ++r) {
                const int m = i * 16 + r + 8 * half;
                const int n = j * 16 + l16;
                outp[(size_t)m * N_DIM + n] = acc[i][j][r];
            }
}

// ---------------------------------------------------------------------------
extern "C" void kernel_launch(void* const* d_in, const int* in_sizes, int n_in,
                              void* d_out, int out_size, void* d_ws, size_t ws_size,
                              hipStream_t stream) {
    const float* x      = (const float*)d_in[0];  // [4096,1024]
    const float* w      = (const float*)d_in[1];  // [1024,4096]
    const float* phases = (const float*)d_in[2];  // [1024,12]
    float* out = (float*)d_out;
    float* pm  = (float*)d_ws;                    // 1024 floats scratch

    qe_phase_mean<<<(K_DIM + 255) / 256, 256, 0, stream>>>(phases, pm, K_DIM);

    dim3 grid(N_DIM / TILE_N, M_DIM / TILE_M);    // 32 x 32 blocks
    qe_wmma_gemm<<<grid, 256, 0, stream>>>(x, w, pm, out);
}
